// AsyncCKConv_71459665871143
// MI455X (gfx1250) — matrix operand order repaired and checked by reading, not compiled, not run on hardware
//
#include <hip/hip_runtime.h>
#include <hip/hip_fp16.h>

typedef __attribute__((ext_vector_type(16))) _Float16 v16h;
typedef __attribute__((ext_vector_type(8)))  _Float16 v8h;
typedef __attribute__((ext_vector_type(8)))  float    v8f;

#define B_ 32
#define N_ 256
#define T_ 128
#define C_ 32
#define H_ 32
#define O_ 64
#define INV_SQRT_2PI 0.3989422804014327f

union HVec16 { v16h v; v8h h[2]; };

// -------- Kernel 1: per-(b,n) Gaussian KDE weight  w_vals = vals / density --------
__global__ __launch_bounds__(N_) void density_kernel(const float* __restrict__ x,
                                                     float* __restrict__ wvals)
{
    __shared__ float s_t[N_];
    __shared__ int   s_f[N_];
    const int b = blockIdx.x;
    const int n = threadIdx.x;
    const float* xr = x + (size_t)(b * N_ + n) * 3;
    const float featf = xr[0], val = xr[1], tim = xr[2];
    s_t[n] = tim;
    s_f[n] = (int)featf;
    __syncthreads();
    const int fn = s_f[n];
    float kd = 0.f;
    int cnt = 0;
    #pragma unroll 4
    for (int i = 0; i < N_; ++i) {
        if (s_f[i] == fn) {
            const float d = tim - s_t[i];
            kd += __expf(-0.5f * d * d);
            ++cnt;
        }
    }
    const float density = (INV_SQRT_2PI * kd) / (float)cnt;
    wvals[b * N_ + n] = val / density;
}

// -------- Kernel 2: one workgroup per (b,t); WMMA for the h1·W2^T matmul --------
__global__ __launch_bounds__(256) void ckconv_kernel(
    const float* __restrict__ x, const float* __restrict__ positions,
    const float* __restrict__ W1, const float* __restrict__ b1, const float* __restrict__ om1p,
    const float* __restrict__ W2, const float* __restrict__ b2, const float* __restrict__ om2p,
    const float* __restrict__ W3, const float* __restrict__ b3, const float* __restrict__ bias,
    const float* __restrict__ wvals, float* __restrict__ out)
{
    __shared__ __align__(16) _Float16 s_h1[N_ * H_];     // 16 KB, row n: 32 halfs
    __shared__ __align__(16) _Float16 s_w2[H_ * H_];     // 2 KB, s_w2[g*32+k] = W2[g][k]
    __shared__ float s_w1[H_ * (C_ + 1)];                // 4224 B, row-major (H, 33)
    __shared__ float s_m[N_];                            // masked w_vals
    __shared__ float s_tdn[N_];
    __shared__ int   s_feat[N_];
    __shared__ float s_b1[H_], s_b2[H_];
    __shared__ float s_s[H_];                            // Σ_n h2[n,g]*m[n]
    __shared__ float s_msum;

    const int t   = blockIdx.x;
    const int b   = blockIdx.y;
    const int tid = threadIdx.x;
    const float om1 = om1p[0];
    const float om2 = om2p[0];
    const float post   = positions[t];
    const float maxpos = positions[T_ - 1];   // positions is sorted ascending

    // ---- stage weights into LDS ----
    for (int i = tid; i < H_ * (C_ + 1); i += 256) s_w1[i] = W1[i];
    for (int i = tid; i < H_ * H_;       i += 256) s_w2[i] = (_Float16)W2[i];
    if (tid < H_) { s_b1[tid] = b1[tid]; s_b2[tid] = b2[tid]; s_s[tid] = 0.f; }
    if (tid == 0) s_msum = 0.f;

    // ---- per-point data: tdn, feat, masked weight ----
    {
        const int n = tid;
        const float* xr = x + (size_t)(b * N_ + n) * 3;
        const float featf = xr[0], val = xr[1], tim = xr[2];
        const bool pad = (featf == 0.f) && (val == 0.f) && (tim == 0.f);
        const float td = tim - post;
        s_tdn[n]  = td / maxpos;
        s_feat[n] = (int)featf;
        s_m[n]    = ((td > 0.f) || pad) ? 0.f : wvals[b * N_ + n];
    }
    __syncthreads();

    // ---- h1 = sin(om1*(W1-gather + tdn*W1[:,C] + b1)) -> LDS f16, row n ----
    {
        const int   n   = tid;
        const int   f   = s_feat[n];
        const float tdn = s_tdn[n];
        #pragma unroll
        for (int h = 0; h < H_; ++h) {
            const float pre = s_w1[h * (C_ + 1) + f] + tdn * s_w1[h * (C_ + 1) + C_] + s_b1[h];
            s_h1[n * H_ + h] = (_Float16)__sinf(om1 * pre);
        }
    }
    atomicAdd(&s_msum, s_m[tid]);
    __syncthreads();

    // ---- WMMA: h2 = sin(om2*(h1 @ W2^T + b2)); fold mask-weight into s[g] ----
    {
        const int wave = tid >> 5;
        const int lane = tid & 31;
        const int l15  = lane & 15;
        const int hi   = lane >> 4;        // 0 or 1 (lane half)

        // B operand (K x N = 32 x 16): lane = column g, K run [hi*16, hi*16+16)
        HVec16 bop[2];
        #pragma unroll
        for (int ct = 0; ct < 2; ++ct) {
            const _Float16* p = &s_w2[(ct * 16 + l15) * H_ + hi * 16];
            bop[ct].h[0] = *(const v8h*)p;
            bop[ct].h[1] = *(const v8h*)(p + 8);
        }

        float acc0 = 0.f, acc1 = 0.f;
        #pragma unroll
        for (int rtl = 0; rtl < 2; ++rtl) {
            const int rt  = wave * 2 + rtl;          // 16-row tile of n
            const int row = rt * 16 + l15;           // A: lane = row M
            const int kb  = hi * 16;                 // A: K run [kb, kb+16)
            HVec16 aop;
            aop.h[0] = *(const v8h*)&s_h1[row * H_ + kb];
            aop.h[1] = *(const v8h*)&s_h1[row * H_ + kb + 8];

            const int nbase = rt * 16 + hi * 8;      // C/D: VGPR r -> M = r + 8*hi
            #pragma unroll
            for (int ct = 0; ct < 2; ++ct) {
                v8f c = {};
                c = __builtin_amdgcn_wmma_f32_16x16x32_f16(
                        /*neg_a=*/false, aop.v, /*neg_b=*/false, bop[ct].v,
                        /*c_mod=*/(short)0, c, /*reuse_a=*/false, /*reuse_b=*/false);
                const float bg = s_b2[ct * 16 + l15];
                float a = 0.f;
                #pragma unroll
                for (int r = 0; r < 8; ++r) {
                    const float h2v = __sinf(om2 * (c[r] + bg));
                    a += h2v * s_m[nbase + r];
                }
                if (ct == 0) acc0 += a; else acc1 += a;
            }
        }
        // lanes l and l+16 cover disjoint rows of the same column g -> plain adds
        atomicAdd(&s_s[l15],      acc0);
        atomicAdd(&s_s[16 + l15], acc1);
    }
    __syncthreads();

    // ---- out[b,o,t] = Σ_g W3[o,g]*s[g] + b3[o]*Σm + bias[o] ----
    if (tid < O_) {
        const int o = tid;
        float acc = b3[o] * s_msum + bias[o];
        #pragma unroll
        for (int g = 0; g < H_; ++g) acc += W3[o * H_ + g] * s_s[g];
        out[((size_t)b * O_ + o) * T_ + t] = acc;
    }
}

extern "C" void kernel_launch(void* const* d_in, const int* in_sizes, int n_in,
                              void* d_out, int out_size, void* d_ws, size_t ws_size,
                              hipStream_t stream) {
    (void)in_sizes; (void)n_in; (void)out_size; (void)ws_size;
    const float* x    = (const float*)d_in[0];
    const float* pos  = (const float*)d_in[1];
    const float* W1   = (const float*)d_in[2];
    const float* b1   = (const float*)d_in[3];
    const float* om1  = (const float*)d_in[4];
    const float* W2   = (const float*)d_in[5];
    const float* b2   = (const float*)d_in[6];
    const float* om2  = (const float*)d_in[7];
    const float* W3   = (const float*)d_in[8];
    const float* b3   = (const float*)d_in[9];
    const float* bias = (const float*)d_in[10];
    float* out   = (float*)d_out;
    float* wvals = (float*)d_ws;               // B_*N_ floats = 32 KB scratch

    density_kernel<<<B_, N_, 0, stream>>>(x, wvals);
    ckconv_kernel<<<dim3(T_, B_), 256, 0, stream>>>(x, pos, W1, b1, om1, W2, b2, om2,
                                                    W3, b3, bias, wvals, out);
}
